// BertSelfAttention_90142773609295
// MI455X (gfx1250) — compile-verified, matrix operand order checked by reading, not run on hardware
//
#include <hip/hip_runtime.h>

#define BS 4
#define SEQ 2048
#define HID 768
#define NH 12
#define HD 64
#define MTOT (BS * SEQ) /* 8192 */

typedef __attribute__((ext_vector_type(16))) __bf16 v16bf;
typedef __attribute__((ext_vector_type(8)))  float  v8f;
typedef __attribute__((ext_vector_type(4)))  unsigned int u32x4;
typedef __attribute__((ext_vector_type(8)))  unsigned int u32x8;
typedef __attribute__((ext_vector_type(2)))  unsigned int u32x2;
typedef __attribute__((ext_vector_type(4)))  float  f32x4;

// ---------------------------------------------------------------- WMMA helper
__device__ inline v8f wmma_bf16(v16bf a, v16bf b, v8f c) {
  // D(f32 16x16) = A(bf16 16x32) * B(bf16 32x16) + C
  return __builtin_amdgcn_wmma_f32_16x16x32_bf16(
      /*neg_a=*/false, a, /*neg_b=*/false, b,
      /*c_mod=*/(short)0, c, /*reuse_a=*/false, /*reuse_b=*/false);
}

// Assemble a 16-element bf16 fragment from two 16-byte chunks.
// Per ISA 16-bit A/B layout: each lane holds K = {kbase..kbase+7} and
// {kbase+16..kbase+23} where kbase = (lane<16)?0:8 — two contiguous 16B runs.
__device__ inline v16bf load_frag(const __bf16* p0, const __bf16* p1) {
  union { u32x4 u[2]; v16bf v; } t;
  t.u[0] = *(const u32x4*)p0;
  t.u[1] = *(const u32x4*)p1;
  return t.v;
}

// ------------------------------------------------- CDNA5 async global -> LDS
// LDS byte address of a __shared__ object = low 32 bits of its generic addr
__device__ inline unsigned lds_off(const void* p) {
  return (unsigned)(size_t)p;
}
// GLOBAL_LOAD_ASYNC_TO_LDS_B128 (tracked by ASYNCcnt): per-lane 16B copy.
__device__ inline void async_copy_b128(unsigned lds_byte_addr, const void* g) {
  asm volatile("global_load_async_to_lds_b128 %0, %1, off"
               :: "v"(lds_byte_addr), "v"((unsigned long long)(size_t)g)
               : "memory");
}
__device__ inline void wait_async() {
  asm volatile("s_wait_asynccnt 0x0" ::: "memory");
}

// ------------------------------------------------- CDNA5 Tensor Data Mover
// 2D tile DMA global->LDS. Tile = tile_d1 rows x tile_d0 8-byte units,
// row stride = stride0 units. Descriptor per CDNA5 ISA.
// Tensor dims set equal to the tile (tile fully in-bounds). One issue per
// wave moves the whole tile (EXEC ignored); tracked by TENSORcnt.
__device__ inline void tdm_load_2d(unsigned lds_addr, const void* gaddr,
                                   unsigned tile_d0, unsigned tile_d1,
                                   unsigned stride0) {
  const unsigned long long ga = (unsigned long long)(size_t)gaddr;
  u32x4 g0;
  g0[0] = 1u;                                    // count=1, user mode
  g0[1] = lds_addr;                              // lds_addr[31:0]
  g0[2] = (unsigned)ga;                          // global_addr[31:0]
  g0[3] = (unsigned)((ga >> 32) & 0x01FFFFFFu)   // global_addr[56:32]
          | (2u << 30);                          // type = 2 ("image")
  u32x8 g1;
  g1[0] = 3u << 16;                              // wg_mask=0, data_size=3 (8B)
  g1[1] = (tile_d0 & 0xFFFFu) << 16;             // tensor_dim0[15:0]
  g1[2] = (tile_d0 >> 16) | (tile_d1 << 16);     // tensor_dim0[31:16], dim1 lo
  g1[3] = (tile_d1 >> 16) | (tile_d0 << 16);     // dim1 hi, tile_dim0
  g1[4] = tile_d1;                               // tile_dim1 (tile_dim2 = 0)
  g1[5] = stride0;                               // tensor_dim0_stride[31:0]
  g1[6] = 0u;
  g1[7] = 0u;
  asm volatile("tensor_load_to_lds %0, %1" :: "s"(g0), "s"(g1) : "memory");
}
__device__ inline void wait_tensor() {
  __builtin_amdgcn_s_wait_tensorcnt((short)0);
}

// ---------------------------------------------------------------- fp32 -> bf16
__global__ __launch_bounds__(256)
void f32_to_bf16_kernel(const float* __restrict__ src, __bf16* __restrict__ dst,
                        int n4) {
  int i = blockIdx.x * blockDim.x + threadIdx.x;
  if (i < n4) {
    f32x4 f = ((const f32x4*)src)[i];
    union { __bf16 e[4]; u32x2 u; } t;
    t.e[0] = (__bf16)f.x; t.e[1] = (__bf16)f.y;
    t.e[2] = (__bf16)f.z; t.e[3] = (__bf16)f.w;
    ((u32x2*)dst)[i] = t.u;
  }
}

// ---------------------------------------------------------------- QKV GEMM
// P[m, n] = sum_k X[m,k] * W[n,k] + bias[n].
// Block = 256 threads = 8 waves; block tile 128(M) x 64(N); wave tile 16x64.
// B tile (64 n-rows x 32 k) staged in LDS with async double-buffering and
// shared by all 8 waves; A fragments come straight from global (per-wave rows).
// VT=false: store [b][h][s][d];  VT=true: store transposed [b][h][d][s].
template <bool VT>
__global__ __launch_bounds__(256)
void qkv_gemm_kernel(const __bf16* __restrict__ X, const __bf16* __restrict__ W,
                     const float* __restrict__ bias, __bf16* __restrict__ out) {
  __shared__ __bf16 Bt[2][64 * 32];    // [buf][n][k] rows of 32 halves (64B)

  const int tid  = threadIdx.x;
  const int lane = tid & 31;
  const int wv   = tid >> 5;
  const int lm   = lane & 15;
  const int hs   = lane >> 4;          // lane half select
  const int kbase = hs ? 8 : 0;
  const int m0 = blockIdx.x * 128 + wv * 16;
  const int n0 = blockIdx.y * 64;

  // staging: thread t copies 16B: B-tile row n = t>>2, k col = (t&3)*8
  const __bf16* wsrc = W + (size_t)(n0 + (tid >> 2)) * HID + (tid & 3) * 8;
  const unsigned sdst[2] = { lds_off(&Bt[0][0]) + (unsigned)tid * 16,
                             lds_off(&Bt[1][0]) + (unsigned)tid * 16 };

  async_copy_b128(sdst[0], wsrc);      // prefetch k-chunk 0

  const __bf16* arow = X + (size_t)(m0 + lm) * HID;
  v8f acc[4] = {};

  #pragma unroll 2
  for (int kc = 0; kc < HID / 32; ++kc) {
    wait_async();                      // staged chunk for this iter is ready
    __syncthreads();                   // visible to all waves; prev buf free
    if (kc + 1 < HID / 32)
      async_copy_b128(sdst[(kc + 1) & 1], wsrc + (kc + 1) * 32);

    const int k0 = kc * 32;
    v16bf a = load_frag(arow + k0 + kbase, arow + k0 + kbase + 16);
    const __bf16* bb = Bt[kc & 1];
    // load all 4 B fragments, then run the WMMA chain
    v16bf bfrag[4];
    #pragma unroll
    for (int jn = 0; jn < 4; ++jn) {
      const __bf16* p = bb + (jn * 16 + lm) * 32 + kbase;
      bfrag[jn] = load_frag(p, p + 16);
    }
    #pragma unroll
    for (int jn = 0; jn < 4; ++jn) acc[jn] = wmma_bf16(a, bfrag[jn], acc[jn]);
  }

  #pragma unroll
  for (int jn = 0; jn < 4; ++jn) {
    const int n = n0 + jn * 16 + lm;
    const float bz = bias[n];
    const int h = n >> 6, d = n & 63;
    #pragma unroll
    for (int r = 0; r < 8; ++r) {
      // C layout: lane<16 -> row r, lane>=16 -> row 8+r; col = lane&15
      int m = m0 + (hs ? 8 + r : r);
      int bbi = m >> 11;               // m / SEQ
      int s   = m & 2047;              // m % SEQ
      float val = acc[jn][r] + bz;
      if (VT)
        out[((size_t)(bbi * NH + h) * HD + d) * SEQ + s] = (__bf16)val;
      else
        out[((size_t)(bbi * NH + h) * SEQ + s) * HD + d] = (__bf16)val;
    }
  }
}

// ---------------------------------------------------------------- Attention
// Flash-style: block = 128 threads = 4 waves; each wave owns 16 q rows.
// K ([s][d]) and V^T ([d][s]) tiles are DMA'd into LDS by the Tensor Data
// Mover (wave 0 issues; TENSORcnt + barrier publishes), double-buffered.
__global__ __launch_bounds__(128)
void attn_kernel(const __bf16* __restrict__ Q, const __bf16* __restrict__ K,
                 const __bf16* __restrict__ VT, const float* __restrict__ mask,
                 float* __restrict__ out) {
  __shared__ __bf16 Kt[2][64 * 64];    // [kv][d]
  __shared__ __bf16 Vt[2][64 * 64];    // [d][kv]
  __shared__ __bf16 Pt[4][16 * 64];    // per-wave P tile, [row][kv]

  const int bh = blockIdx.y;
  const int b = bh / NH, h = bh % NH;
  const int tid  = threadIdx.x;
  const int wv   = tid >> 5;
  const int lane = tid & 31;
  const int lm   = lane & 15;
  const int hs   = lane >> 4;
  const int kbase = hs ? 8 : 0;
  const int q0 = blockIdx.x * 64 + wv * 16;

  const size_t head = (size_t)(b * NH + h) * SEQ;
  const __bf16* Qg  = Q  + head * HD;  // [s][d]
  const __bf16* Kg  = K  + head * HD;  // [s][d]
  const __bf16* VTg = VT + head * HD;  // [d][s] (same flat base)
  const float*  mrow = mask + (size_t)b * SEQ;

  // Preload Q fragments (A-matrix, rows = q, K-dim = d, 2 steps of 32)
  v16bf aq[2];
  #pragma unroll
  for (int kk = 0; kk < 2; ++kk) {
    const __bf16* p = Qg + (size_t)(q0 + lm) * HD + kk * 32 + kbase;
    aq[kk] = load_frag(p, p + 16);
  }

  v8f o[4] = {};                       // 16 x 64 fp32 accumulator (4 d-tiles)
  float m_i[8], l_i[8];
  #pragma unroll
  for (int r = 0; r < 8; ++r) { m_i[r] = -3.0e38f; l_i[r] = 0.0f; }

  // TDM stage of one 64-key chunk: K tile 64 rows x 16 units (stride 16),
  // V^T tile 64 rows x 16 units (stride SEQ*2/8 = 512 units).
  auto stage = [&](int buf, int kv0) {
    if (wv == 0) {
      tdm_load_2d(lds_off(&Kt[buf][0]), Kg + (size_t)kv0 * HD, 16, 64, 16);
      tdm_load_2d(lds_off(&Vt[buf][0]), VTg + kv0,            16, 64, 512);
    }
  };

  stage(0, 0);

  #pragma unroll 1
  for (int it = 0; it < SEQ / 64; ++it) {
    const int cur = it & 1;
    const int kv0 = it * 64;
    if (wv == 0) wait_tensor();        // this iter's tiles landed in LDS
    __syncthreads();                   // publish to all waves; other buf free
    if (it + 1 < SEQ / 64) stage(cur ^ 1, kv0 + 64);

    const __bf16* kcur = Kt[cur];
    const __bf16* vcur = Vt[cur];

    // ---- S = Q * K^T: load all 8 B fragments, then the WMMA chain
    v16bf bk[4][2];
    #pragma unroll
    for (int j = 0; j < 4; ++j) {
      #pragma unroll
      for (int kk = 0; kk < 2; ++kk) {
        const __bf16* p = kcur + (size_t)(j * 16 + lm) * 64 + kk * 32 + kbase;
        bk[j][kk] = load_frag(p, p + 16);
      }
    }
    v8f s[4];
    #pragma unroll
    for (int j = 0; j < 4; ++j) {
      v8f acc = {};
      acc = wmma_bf16(aq[0], bk[j][0], acc);
      acc = wmma_bf16(aq[1], bk[j][1], acc);
      s[j] = acc;
    }

    // ---- scale 1/sqrt(64) + additive mask (mask < -1 -> -1e9)
    float mv[4];
    #pragma unroll
    for (int j = 0; j < 4; ++j) mv[j] = mrow[kv0 + j * 16 + lm];
    #pragma unroll
    for (int j = 0; j < 4; ++j) {
      const bool bad = mv[j] < -1.0f;
      #pragma unroll
      for (int r = 0; r < 8; ++r) s[j][r] = bad ? -1.0e9f : s[j][r] * 0.125f;
    }

    // ---- online softmax; rows on VGPR index, cols across 16-lane halves
    #pragma unroll
    for (int r = 0; r < 8; ++r) {
      float mx = fmaxf(fmaxf(s[0][r], s[1][r]), fmaxf(s[2][r], s[3][r]));
      #pragma unroll
      for (int off = 8; off > 0; off >>= 1) mx = fmaxf(mx, __shfl_xor(mx, off, 32));
      float mnew = fmaxf(m_i[r], mx);
      float sc = __expf(m_i[r] - mnew);
      m_i[r] = mnew;
      l_i[r] *= sc;
      #pragma unroll
      for (int j = 0; j < 4; ++j) o[j][r] *= sc;
      float ps = 0.0f;
      #pragma unroll
      for (int j = 0; j < 4; ++j) {
        float p = __expf(s[j][r] - mnew);
        s[j][r] = p;
        ps += p;
      }
      #pragma unroll
      for (int off = 8; off > 0; off >>= 1) ps += __shfl_xor(ps, off, 32);
      l_i[r] += ps;
    }

    // ---- P (C layout) -> LDS bf16 so it can be re-read in A layout
    __bf16* pw = Pt[wv];
    #pragma unroll
    for (int j = 0; j < 4; ++j) {
      #pragma unroll
      for (int r = 0; r < 8; ++r) {
        int row = hs ? 8 + r : r;
        pw[row * 64 + j * 16 + lm] = (__bf16)s[j][r];
      }
    }

    // ---- O += P * V: load all fragments, then the WMMA chain
    v16bf ap[2];
    v16bf bv_[2][4];
    #pragma unroll
    for (int kk = 0; kk < 2; ++kk) {
      const __bf16* pa = pw + (size_t)lm * 64 + kk * 32 + kbase;
      ap[kk] = load_frag(pa, pa + 16);
      #pragma unroll
      for (int jd = 0; jd < 4; ++jd) {
        const __bf16* pb = vcur + (size_t)(jd * 16 + lm) * 64 + kk * 32 + kbase;
        bv_[kk][jd] = load_frag(pb, pb + 16);
      }
    }
    #pragma unroll
    for (int kk = 0; kk < 2; ++kk)
      #pragma unroll
      for (int jd = 0; jd < 4; ++jd)
        o[jd] = wmma_bf16(ap[kk], bv_[kk][jd], o[jd]);
  }

  // ---- normalize and write fp32 output [b][s][h*64+d]
  #pragma unroll
  for (int r = 0; r < 8; ++r) {
    float inv = 1.0f / l_i[r];
    int row = q0 + (hs ? 8 + r : r);
    float* op = out + ((size_t)b * SEQ + row) * HID + h * HD;
    #pragma unroll
    for (int jd = 0; jd < 4; ++jd) op[jd * 16 + lm] = o[jd][r] * inv;
  }
}

// ---------------------------------------------------------------- launcher
extern "C" void kernel_launch(void* const* d_in, const int* in_sizes, int n_in,
                              void* d_out, int out_size, void* d_ws, size_t ws_size,
                              hipStream_t stream) {
  const float* hsx  = (const float*)d_in[0];
  const float* mask = (const float*)d_in[1];
  const float* Wq   = (const float*)d_in[2];
  const float* bq   = (const float*)d_in[3];
  const float* Wk   = (const float*)d_in[4];
  const float* bk   = (const float*)d_in[5];
  const float* Wv   = (const float*)d_in[6];
  const float* bv   = (const float*)d_in[7];
  float* out = (float*)d_out;

  // workspace layout (bf16 buffers)
  constexpr size_t XB_BYTES = (size_t)MTOT * HID * 2;         // 12.6 MB
  constexpr size_t WB_BYTES = (size_t)HID * HID * 2;          // 1.18 MB
  constexpr size_t HB_BYTES = (size_t)BS * NH * SEQ * HD * 2; // 12.6 MB
  char* ws = (char*)d_ws;
  __bf16* Xb  = (__bf16*)(ws);
  __bf16* Wqb = (__bf16*)(ws + XB_BYTES);
  __bf16* Wkb = (__bf16*)(ws + XB_BYTES + WB_BYTES);
  __bf16* Wvb = (__bf16*)(ws + XB_BYTES + 2 * WB_BYTES);
  __bf16* Qb  = (__bf16*)(ws + XB_BYTES + 3 * WB_BYTES);
  __bf16* Kb  = (__bf16*)(ws + XB_BYTES + 3 * WB_BYTES + HB_BYTES);
  __bf16* Vb  = (__bf16*)(ws + XB_BYTES + 3 * WB_BYTES + 2 * HB_BYTES);

  // fp32 -> bf16 conversions
  {
    int n4 = MTOT * HID / 4;
    f32_to_bf16_kernel<<<(n4 + 255) / 256, 256, 0, stream>>>(hsx, Xb, n4);
  }
  {
    int n4 = HID * HID / 4;
    int g = (n4 + 255) / 256;
    f32_to_bf16_kernel<<<g, 256, 0, stream>>>(Wq, Wqb, n4);
    f32_to_bf16_kernel<<<g, 256, 0, stream>>>(Wk, Wkb, n4);
    f32_to_bf16_kernel<<<g, 256, 0, stream>>>(Wv, Wvb, n4);
  }

  // QKV projections (WMMA bf16); V is stored transposed [b][h][d][s]
  dim3 gg(MTOT / 128, HID / 64);
  qkv_gemm_kernel<false><<<gg, 256, 0, stream>>>(Xb, Wqb, bq, Qb);
  qkv_gemm_kernel<false><<<gg, 256, 0, stream>>>(Xb, Wkb, bk, Kb);
  qkv_gemm_kernel<true ><<<gg, 256, 0, stream>>>(Xb, Wvb, bv, Vb);

  // Fused flash attention (WMMA bf16, fp32 softmax/accum, TDM staging)
  attn_kernel<<<dim3(SEQ / 64, BS * NH), 128, 0, stream>>>(Qb, Kb, Vb, mask, out);
}
